// PANPool_49228915147363
// MI455X (gfx1250) — matrix operand order repaired and checked by reading, not compiled, not run on hardware
//
#include <hip/hip_runtime.h>
#include <hip/hip_bf16.h>

typedef __attribute__((ext_vector_type(16))) _Float16 v16h;
typedef __attribute__((ext_vector_type(8)))  _Float16 v8h;
typedef __attribute__((ext_vector_type(8)))  float    v8f;

static inline int cdiv(long a, int b) { return (int)((a + b - 1) / b); }

__device__ __forceinline__ v16h mk16(v8h lo, v8h hi)
{
    v16h r;
#pragma unroll
    for (int i = 0; i < 8; ++i) { r[i] = lo[i]; r[8 + i] = hi[i]; }
    return r;
}

// ---------------------------------------------------------------------------
// NT GEMM via WMMA f16->f32:  C[b] (MxN) = A[b] (MxK row-major) * BT[b] (NxK)^T
// block = 128 threads = 4 waves; wave computes 16 rows x 64 cols (4 wmma tiles).
// Epilogues: +bias[col] & relu -> Cf (fp32); f16 copy -> Ch;
//            fused power accumulation Macc[off] += (wvec[0]*wvec[wi]) * v.
// ---------------------------------------------------------------------------
__global__ __launch_bounds__(128)
void k_gemm_nt(const _Float16* __restrict__ A, const _Float16* __restrict__ BT,
               float* __restrict__ Cf, _Float16* __restrict__ Ch,
               float* __restrict__ Macc, const float* __restrict__ wvec, int wi,
               const float* __restrict__ bias, int relu,
               int Ndim, int Kdim,
               long sA, long sBT, long sC)
{
    const int b     = blockIdx.z;
    const int wid   = threadIdx.x >> 5;
    const int lane  = threadIdx.x & 31;
    const int lhalf = lane >> 4;   // 0: lanes 0-15, 1: lanes 16-31
    const int l16   = lane & 15;
    const int m0 = blockIdx.y * 64 + wid * 16;
    const int n0 = blockIdx.x * 64;

    v8f acc0, acc1, acc2, acc3;
#pragma unroll
    for (int r = 0; r < 8; ++r) { acc0[r] = 0.0f; acc1[r] = 0.0f; acc2[r] = 0.0f; acc3[r] = 0.0f; }

    // A fragment: lanes 0-15 -> K {0..15}, lanes 16-31 -> K {16..31} (matches B)
    const _Float16* ap = A + (long)b * sA + (long)(m0 + l16) * Kdim + (lhalf ? 16 : 0);
    // B fragment: lane = column, 16 contiguous K values (0-15 / 16-31 halves).
    // Four scalar pointer PHIs (NOT an array) so addrspace inference keeps these global_load.
    const _Float16* bp0 = BT + (long)b * sBT + (long)(n0 + l16) * Kdim + (lhalf ? 16 : 0);
    const _Float16* bp1 = bp0 + 16 * (long)Kdim;
    const _Float16* bp2 = bp1 + 16 * (long)Kdim;
    const _Float16* bp3 = bp2 + 16 * (long)Kdim;

    for (int k0 = 0; k0 < Kdim; k0 += 32) {
        // issue every load of this K-step before any WMMA consumes one
        v8h alo = *(const v8h*)(ap);
        v8h ahi = *(const v8h*)(ap + 8);
        v8h b0l = *(const v8h*)(bp0), b0h = *(const v8h*)(bp0 + 8);
        v8h b1l = *(const v8h*)(bp1), b1h = *(const v8h*)(bp1 + 8);
        v8h b2l = *(const v8h*)(bp2), b2h = *(const v8h*)(bp2 + 8);
        v8h b3l = *(const v8h*)(bp3), b3h = *(const v8h*)(bp3 + 8);
        __builtin_prefetch(ap + 32, 0, 3);   // speculative: silently dropped if OOB
        ap += 32; bp0 += 32; bp1 += 32; bp2 += 32; bp3 += 32;

        v16h af = mk16(alo, ahi);
        acc0 = __builtin_amdgcn_wmma_f32_16x16x32_f16(
            false, af, false, mk16(b0l, b0h), (short)0, acc0, false, false);
        acc1 = __builtin_amdgcn_wmma_f32_16x16x32_f16(
            false, af, false, mk16(b1l, b1h), (short)0, acc1, false, false);
        acc2 = __builtin_amdgcn_wmma_f32_16x16x32_f16(
            false, af, false, mk16(b2l, b2h), (short)0, acc2, false, false);
        acc3 = __builtin_amdgcn_wmma_f32_16x16x32_f16(
            false, af, false, mk16(b3l, b3h), (short)0, acc3, false, false);
    }

    const float coef = (Macc != nullptr) ? (wvec[0] * wvec[wi]) : 0.0f;
    v8f accs[4] = {acc0, acc1, acc2, acc3};
#pragma unroll
    for (int j = 0; j < 4; ++j) {
        const int col = n0 + j * 16 + l16;
        const float bv = bias ? bias[col] : 0.0f;
#pragma unroll
        for (int r = 0; r < 8; ++r) {
            const int row = m0 + r + lhalf * 8;   // C layout: VGPR r -> M=r / r+8
            float v = accs[j][r] + bv;
            if (relu) v = v > 0.0f ? v : 0.0f;
            const long off = (long)b * sC + (long)row * Ndim + col;
            if (Cf) Cf[off] = v;
            if (Ch) Ch[off] = (_Float16)v;
            if (Macc) Macc[off] += coef * v;      // tile owned by this wave: no race
        }
    }
}

// ----------------------------- elementwise ---------------------------------
__global__ void k_convert(const float* __restrict__ s, _Float16* __restrict__ d, long total)
{
    for (long i = blockIdx.x * (long)blockDim.x + threadIdx.x; i < total;
         i += (long)gridDim.x * blockDim.x)
        d[i] = (_Float16)s[i];
}

// dst[b][c][r] = src[b][r][c]
__global__ void k_transpose(const float* __restrict__ src, _Float16* __restrict__ dst,
                            int R, int C, long sS, long sD, long total)
{
    for (long i = blockIdx.x * (long)blockDim.x + threadIdx.x; i < total;
         i += (long)gridDim.x * blockDim.x) {
        long b = i / ((long)R * C);
        long rem = i - b * (long)R * C;
        int r = (int)(rem / C), c = (int)(rem - (long)r * C);
        dst[b * sD + (long)c * R + r] = (_Float16)src[b * sS + (long)r * C + c];
    }
}

// M = w0*w1*A + w0*I
__global__ void k_minit(const float* __restrict__ A, float* __restrict__ M,
                        const float* __restrict__ w, int n, long total)
{
    const float w0 = w[0], c = w[0] * w[1];
    for (long i = blockIdx.x * (long)blockDim.x + threadIdx.x; i < total;
         i += (long)gridDim.x * blockDim.x) {
        long rem = i % ((long)n * n);
        int r = (int)(rem / n), cc = (int)(rem - (long)r * n);
        M[i] = c * A[i] + ((r == cc) ? w0 : 0.0f);
    }
}

// one wave per row: dinv = rsqrt(#nonzeros)
__global__ __launch_bounds__(128)
void k_rowdeg(const float* __restrict__ M, float* __restrict__ dinv, int n)
{
    const int w = (int)((blockIdx.x * (long)blockDim.x + threadIdx.x) >> 5); // b*n + row
    const int lane = threadIdx.x & 31;
    const float* row = M + (long)w * n;
    float cnt = 0.0f;
    for (int c = lane; c < n; c += 32) cnt += (row[c] != 0.0f) ? 1.0f : 0.0f;
#pragma unroll
    for (int o = 16; o > 0; o >>= 1) cnt += __shfl_xor(cnt, o, 32);
    if (lane == 0) dinv[w] = (cnt > 0.0f) ? rsqrtf(cnt) : 0.0f;
}

// M = dinv_r * M * dinv_c  (in place), also f16 copy for WMMA
__global__ void k_normalize(float* __restrict__ M, _Float16* __restrict__ Mh,
                            const float* __restrict__ dinv, int n, long total)
{
    for (long i = blockIdx.x * (long)blockDim.x + threadIdx.x; i < total;
         i += (long)gridDim.x * blockDim.x) {
        long b = i / ((long)n * n);
        long rem = i - b * (long)n * n;
        int r = (int)(rem / n), c = (int)(rem - (long)r * n);
        float v = M[i] * dinv[b * n + r] * dinv[b * n + c];
        M[i] = v; Mh[i] = (_Float16)v;
    }
}

__global__ void k_colsum(const float* __restrict__ M, float* __restrict__ cs, int n)
{
    long i = blockIdx.x * (long)blockDim.x + threadIdx.x;
    if (i >= 4L * n) return;
    long b = i / n; int c = (int)(i - b * n);
    const float* base = M + b * (long)n * n + c;
    float s = 0.0f;
    for (int r = 0; r < n; ++r) s += base[(long)r * n];
    cs[i] = s;
}

// one wave per node: score = tanh(b0*<x,p> + b1*colsum)
__global__ __launch_bounds__(128)
void k_score(const float* __restrict__ X, const float* __restrict__ p,
             const float* __restrict__ beta, const float* __restrict__ cs,
             float* __restrict__ score, int d)
{
    const int w = (int)((blockIdx.x * (long)blockDim.x + threadIdx.x) >> 5); // b*n+i
    const int lane = threadIdx.x & 31;
    const float* xr = X + (long)w * d;
    float s = 0.0f;
    for (int f = lane; f < d; f += 32) s += xr[f] * p[f];
#pragma unroll
    for (int o = 16; o > 0; o >>= 1) s += __shfl_xor(s, o, 32);
    if (lane == 0) score[w] = tanhf(beta[0] * s + beta[1] * cs[w]);
}

// per-batch bitonic sort (descending, tie -> lower index first), emit top-k
__global__ void k_topk(const float* __restrict__ score, int* __restrict__ idx,
                       float* __restrict__ vals, int n, int k)
{
    __shared__ float ss[1024];
    __shared__ int   si[1024];
    const int b = blockIdx.x, t = threadIdx.x;
    ss[t] = score[(long)b * n + t]; si[t] = t;
    __syncthreads();
    for (int size = 2; size <= n; size <<= 1) {
        for (int stride = size >> 1; stride > 0; stride >>= 1) {
            const int j = t ^ stride;
            if (j > t) {
                const float st = ss[t], sj = ss[j];
                const int   it = si[t], ij = si[j];
                const bool before_tj = (st > sj) || (st == sj && it < ij);
                const bool up = ((t & size) == 0);
                if (up ? !before_tj : before_tj) {
                    ss[t] = sj; ss[j] = st; si[t] = ij; si[j] = it;
                }
            }
            __syncthreads();
        }
    }
    if (t < k) { idx[(long)b * k + t] = si[t]; vals[(long)b * k + t] = ss[t]; }
}

__global__ void k_gather_x(const float* __restrict__ X, const int* __restrict__ idx,
                           const float* __restrict__ vals, float* __restrict__ xo,
                           int n, int k, int d, long total)
{
    for (long i = blockIdx.x * (long)blockDim.x + threadIdx.x; i < total;
         i += (long)gridDim.x * blockDim.x) {
        long b = i / ((long)k * d);
        long rem = i - b * (long)k * d;
        int t = (int)(rem / d), f = (int)(rem - (long)t * d);
        int g = idx[b * k + t];
        xo[i] = X[(b * (long)n + g) * d + f] * vals[b * k + t];
    }
}

__global__ void k_gather_A(const float* __restrict__ M, const int* __restrict__ idx,
                           float* __restrict__ Ao, int n, int k, long total)
{
    for (long i = blockIdx.x * (long)blockDim.x + threadIdx.x; i < total;
         i += (long)gridDim.x * blockDim.x) {
        long b = i / ((long)k * k);
        long rem = i - b * (long)k * k;
        int r = (int)(rem / k), c = (int)(rem - (long)r * k);
        float v = M[b * (long)n * n + (long)idx[b * k + r] * n + idx[b * k + c]];
        Ao[i] = (v != 0.0f) ? 1.0f : 0.0f;
    }
}

// mean-pool (128 nodes x 256 feat) -> fc 256->128 relu -> 128->64 relu -> 64->10
__global__ __launch_bounds__(256)
void k_head(const float* __restrict__ X,
            const float* __restrict__ w1, const float* __restrict__ b1,
            const float* __restrict__ w2, const float* __restrict__ b2,
            const float* __restrict__ w3, const float* __restrict__ b3,
            float* __restrict__ out, int nNodes)
{
    __shared__ float g[256], h1[128], h2[64];
    const int b = blockIdx.x, t = threadIdx.x;
    float s = 0.0f;
    for (int i = 0; i < nNodes; ++i) s += X[((long)b * nNodes + i) * 256 + t];
    g[t] = s / (float)nNodes;
    __syncthreads();
    if (t < 128) {
        float a = b1[t];
        for (int j = 0; j < 256; ++j) a += w1[t * 256 + j] * g[j];
        h1[t] = a > 0.0f ? a : 0.0f;
    }
    __syncthreads();
    if (t < 64) {
        float a = b2[t];
        for (int j = 0; j < 128; ++j) a += w2[t * 128 + j] * h1[j];
        h2[t] = a > 0.0f ? a : 0.0f;
    }
    __syncthreads();
    if (t < 10) {
        float a = b3[t];
        for (int j = 0; j < 64; ++j) a += w3[t * 64 + j] * h2[j];
        out[b * 10 + t] = a;
    }
}

// ---------------------------------------------------------------------------
extern "C" void kernel_launch(void* const* d_in, const int* in_sizes, int n_in,
                              void* d_out, int out_size, void* d_ws, size_t ws_size,
                              hipStream_t stream)
{
    (void)in_sizes; (void)n_in; (void)out_size; (void)ws_size;
    const int B = 4, N0 = 1024, HID = 256;
    const float* x_in = (const float*)d_in[0];
    const float* A_in = (const float*)d_in[1];
    const float *W[3], *LW[3], *LB[3], *PP[3], *BT[3];
    for (int l = 0; l < 3; ++l) {
        W[l]  = (const float*)d_in[2 + 5 * l];
        LW[l] = (const float*)d_in[3 + 5 * l];
        LB[l] = (const float*)d_in[4 + 5 * l];
        PP[l] = (const float*)d_in[5 + 5 * l];
        BT[l] = (const float*)d_in[6 + 5 * l];
    }
    const float* fc1w = (const float*)d_in[17];
    const float* fc1b = (const float*)d_in[18];
    const float* fc2w = (const float*)d_in[19];
    const float* fc2b = (const float*)d_in[20];
    const float* fc3w = (const float*)d_in[21];
    const float* fc3b = (const float*)d_in[22];
    float* out = (float*)d_out;

    char* wp = (char*)d_ws;
    auto carve = [&](size_t bytes) {
        char* r = wp; wp += (bytes + 255) & ~(size_t)255; return (void*)r;
    };
    float*    Mbuf  = (float*)   carve((size_t)B * N0 * N0 * 4);
    _Float16* ATh   = (_Float16*)carve((size_t)B * N0 * N0 * 2);
    _Float16* Pa    = (_Float16*)carve((size_t)B * N0 * N0 * 2);
    _Float16* Pb    = (_Float16*)carve((size_t)B * N0 * N0 * 2);
    _Float16* Mh    = (_Float16*)carve((size_t)B * N0 * N0 * 2);
    _Float16* xTh   = (_Float16*)carve((size_t)B * HID * N0 * 2);
    _Float16* lWh   = (_Float16*)carve((size_t)HID * HID * 2);
    _Float16* Hh    = (_Float16*)carve((size_t)B * N0 * HID * 2);
    float*    Xf    = (float*)   carve((size_t)B * N0 * HID * 4);
    float*    xnext = (float*)   carve((size_t)B * (N0 / 2) * HID * 4);
    float*    Anext = (float*)   carve((size_t)B * (N0 / 2) * (N0 / 2) * 4);
    float*    dinv  = (float*)   carve((size_t)B * N0 * 4);
    float*    csum  = (float*)   carve((size_t)B * N0 * 4);
    float*    scr   = (float*)   carve((size_t)B * N0 * 4);
    float*    vals  = (float*)   carve((size_t)B * N0 * 4);
    int*      idxb  = (int*)     carve((size_t)B * N0 * 4);

    const float* xsrc = x_in;
    const float* Asrc = A_in;
    int n = N0, din = 128;

    for (int l = 0; l < 3; ++l) {
        const long nn = (long)B * n * n;
        // A -> ATh (f16, transposed for NT-GEMM B operand) and Pa = f16(A) = A^1
        k_transpose<<<cdiv(nn, 256), 256, 0, stream>>>(Asrc, ATh, n, n,
                                                       (long)n * n, (long)n * n, nn);
        k_convert<<<cdiv(nn, 256), 256, 0, stream>>>(Asrc, Pa, nn);
        // M = w0*I + w0*w1*A
        k_minit<<<cdiv(nn, 256), 256, 0, stream>>>(Asrc, Mbuf, W[l], n, nn);
        // powers A^2..A^4: WMMA with fused epilogue
        //   Pn = f16(P @ A)   and   M += (w0*w_i) * (P @ A)
        // (exact: f16 integer inputs, f32 accumulate; structure preserved)
        _Float16* Pc = Pa; _Float16* Pn = Pb;
        for (int i = 2; i <= 4; ++i) {
            dim3 g(n / 64, n / 64, B);
            k_gemm_nt<<<g, 128, 0, stream>>>(Pc, ATh, nullptr, Pn, Mbuf, W[l], i,
                                             nullptr, 0, n, n,
                                             (long)n * n, (long)n * n, (long)n * n);
            _Float16* t = Pc; Pc = Pn; Pn = t;
        }
        // degree (structural nnz) -> rsqrt, then symmetric normalize (+f16 copy)
        k_rowdeg<<<(B * n) / 4, 128, 0, stream>>>(Mbuf, dinv, n);
        k_normalize<<<cdiv(nn, 256), 256, 0, stream>>>(Mbuf, Mh, dinv, n, nn);
        // H = Mnorm @ x  (WMMA, f16 out)
        const long nx = (long)B * n * din;
        k_transpose<<<cdiv(nx, 256), 256, 0, stream>>>(xsrc, xTh, n, din,
                                                       (long)n * din, (long)din * n, nx);
        {
            dim3 g(din / 64, n / 64, B);
            k_gemm_nt<<<g, 128, 0, stream>>>(Mh, xTh, nullptr, Hh, nullptr, nullptr, 0,
                                             nullptr, 0, din, n,
                                             (long)n * n, (long)din * n, (long)n * din);
        }
        // X = relu(H @ lin_w^T + b): lin_w row-major IS the NT B-operand
        k_convert<<<cdiv((long)HID * din, 256), 256, 0, stream>>>(LW[l], lWh,
                                                                  (long)HID * din);
        {
            dim3 g(HID / 64, n / 64, B);
            k_gemm_nt<<<g, 128, 0, stream>>>(Hh, lWh, Xf, nullptr, nullptr, nullptr, 0,
                                             LB[l], 1, HID, din,
                                             (long)n * din, 0L, (long)n * HID);
        }
        // pooling
        k_colsum<<<cdiv((long)B * n, 256), 256, 0, stream>>>(Mbuf, csum, n);
        k_score<<<(B * n) / 4, 128, 0, stream>>>(Xf, PP[l], BT[l], csum, scr, HID);
        const int k = n / 2;
        k_topk<<<B, n, 0, stream>>>(scr, idxb, vals, n, k);
        const long ngx = (long)B * k * HID;
        k_gather_x<<<cdiv(ngx, 256), 256, 0, stream>>>(Xf, idxb, vals, xnext,
                                                       n, k, HID, ngx);
        const long nga = (long)B * k * k;
        k_gather_A<<<cdiv(nga, 256), 256, 0, stream>>>(Mbuf, idxb, Anext, n, k, nga);
        xsrc = xnext; Asrc = Anext; n = k; din = HID;
    }
    // n == 128 here
    k_head<<<B, 256, 0, stream>>>(xsrc, fc1w, fc1b, fc2w, fc2b, fc3w, fc3b, out, n);
}
